// DeepMultiOmicNetV3_39436389712065
// MI455X (gfx1250) — matrix operand order repaired and checked by reading, not compiled, not run on hardware
//
#include <hip/hip_runtime.h>
#include <hip/hip_bf16.h>

// ---------------------------------------------------------------------------
// Problem constants (from reference)
// ---------------------------------------------------------------------------
#define BATCH      1024
#define IN_DIM     5000
#define IN_PAD     5056     // padded to multiple of 64 for BK=64 staging
#define NUM_OMICS  3
#define HIDDEN_W   2000
#define KH_PAD     2048     // hidden width padded to multiple of 64 (as K)
#define OUT_DIM    200
#define HIDDEN_SIZE 2
#define GROUP      20
#define GW         100
#define GW_PAD     128      // per-group K padded (multiple of 64)

typedef unsigned short u16;
typedef __attribute__((ext_vector_type(8)))  u16    u16x8;
typedef __attribute__((ext_vector_type(16))) __bf16 v16bf;
typedef __attribute__((ext_vector_type(8)))  float  v8f;

struct FragBits { u16x8 lo; u16x8 hi; };

// ---------------------------------------------------------------------------
// Helpers
// ---------------------------------------------------------------------------
__device__ __forceinline__ u16 f2bf(float f) {
    unsigned u = __builtin_bit_cast(unsigned, f);
    unsigned r = (u + 0x7FFFu + ((u >> 16) & 1u)) >> 16;   // round-nearest-even
    return (u16)r;
}
__device__ __forceinline__ float bf2f(u16 h) {
    unsigned u = ((unsigned)h) << 16;
    return __builtin_bit_cast(float, u);
}
__device__ __forceinline__ float sigmoidf_(float x) {
    return 1.0f / (1.0f + __expf(-x));
}
__device__ __forceinline__ v16bf make_frag(const u16* p0, const u16* p1) {
    FragBits t;
    t.lo = *(const u16x8*)p0;
    t.hi = *(const u16x8*)p1;
    return __builtin_bit_cast(v16bf, t);
}

// CDNA5 async copy: 16 bytes/lane, global -> LDS, tracked by ASYNCcnt.
__device__ __forceinline__ void async_load_b128(unsigned lds_off, const void* gptr) {
    asm volatile("global_load_async_to_lds_b128 %0, %1, off"
                 :: "v"(lds_off), "v"(gptr)
                 : "memory");
}
__device__ __forceinline__ void wait_async0() {
    asm volatile("s_wait_asynccnt 0x0" ::: "memory");
}

// ---------------------------------------------------------------------------
// Elementwise / conversion kernels
// ---------------------------------------------------------------------------
__global__ __launch_bounds__(256) void merge_kernel(
    const float* __restrict__ x, const float* __restrict__ mw,
    const float* __restrict__ mb, u16* __restrict__ m)
{
    int idx = blockIdx.x * 256 + threadIdx.x;
    if (idx >= BATCH * IN_PAD) return;
    int b = idx / IN_PAD;
    int i = idx - b * IN_PAD;
    if (i >= IN_DIM) { m[idx] = 0; return; }
    const float* xp = x + ((size_t)b * IN_DIM + i) * NUM_OMICS;
    float v = xp[0] * mw[i * 3 + 0] + xp[1] * mw[i * 3 + 1] + xp[2] * mw[i * 3 + 2] + mb[i];
    m[idx] = f2bf(sigmoidf_(v));
}

// h0_w [2000,5000] f32 -> bf16 [2000, IN_PAD] zero padded
__global__ __launch_bounds__(256) void conv_h0w_kernel(
    const float* __restrict__ w, u16* __restrict__ o)
{
    int idx = blockIdx.x * 256 + threadIdx.x;
    if (idx >= HIDDEN_W * IN_PAD) return;
    int n = idx / IN_PAD;
    int k = idx - n * IN_PAD;
    o[idx] = (k < IN_DIM) ? f2bf(w[(size_t)n * IN_DIM + k]) : (u16)0;
}

// bn_w1 [L,G,H,GW] f32 -> w1t bf16 [L, n=g*100+j, h(pad 2048)]
__global__ __launch_bounds__(256) void conv_w1_kernel(
    const float* __restrict__ w, u16* __restrict__ o)
{
    long long idx = (long long)blockIdx.x * 256 + threadIdx.x;
    if (idx >= (long long)HIDDEN_SIZE * HIDDEN_W * KH_PAD) return;
    int h   = (int)(idx % KH_PAD);
    int n   = (int)((idx / KH_PAD) % HIDDEN_W);
    int l   = (int)(idx / ((long long)KH_PAD * HIDDEN_W));
    if (h >= HIDDEN_W) { o[idx] = 0; return; }
    int g = n / GW, j = n - g * GW;
    o[idx] = f2bf(w[(((size_t)l * GROUP + g) * HIDDEN_W + h) * GW + j]);
}

// bn_w2 [L,G,GW,GW] f32 -> w2t bf16 [L,G, j, k(pad 128)]
__global__ __launch_bounds__(256) void conv_w2_kernel(
    const float* __restrict__ w, u16* __restrict__ o)
{
    int idx = blockIdx.x * 256 + threadIdx.x;
    if (idx >= HIDDEN_SIZE * GROUP * GW * GW_PAD) return;
    int k  = idx & (GW_PAD - 1);
    int j  = (idx / GW_PAD) % GW;
    int g  = (idx / (GW_PAD * GW)) % GROUP;
    int l  = idx / (GW_PAD * GW * GROUP);
    o[idx] = (k < GW) ? f2bf(w[(((size_t)l * GROUP + g) * GW + k) * GW + j]) : (u16)0;
}

// out_w [200,2000] f32 -> bf16 [200, KH_PAD] zero padded
__global__ __launch_bounds__(256) void conv_outw_kernel(
    const float* __restrict__ w, u16* __restrict__ o)
{
    int idx = blockIdx.x * 256 + threadIdx.x;
    if (idx >= OUT_DIM * KH_PAD) return;
    int n = idx / KH_PAD;
    int k = idx - n * KH_PAD;
    o[idx] = (k < HIDDEN_W) ? f2bf(w[(size_t)n * HIDDEN_W + k]) : (u16)0;
}

__global__ __launch_bounds__(256) void zero_u16_kernel(u16* __restrict__ p, int n)
{
    int idx = blockIdx.x * 256 + threadIdx.x;
    if (idx < n) p[idx] = 0;
}

// ---------------------------------------------------------------------------
// Generic bf16 WMMA GEMM:  C[M,N] = epilogue( A[M,K] * B[N,K]^T + bias )
//   block tile 128x128, BK=64 (two WMMA k-substeps per stage), 256 thr / 8 waves
//   Double-buffered LDS fed by global_load_async_to_lds_b128 (ASYNCcnt):
//   one s_wait_asynccnt + one barrier per 16 WMMAs. LDS pre-zeroed once;
//   OOB rows never written (exec-masked issue) so they stay zero. K % 64 == 0.
//   Interior tiles take a branchless unconditional async-issue path.
// Epilogues:
//   0: f32 out, += bias                         (output layer)
//   1: sigmoid -> bf16 out                      (h0)
//   2: sigmoid -> bf16, col -> g*128 + (col%100) group-padded  (bottleneck g1)
//   3: t=sigmoid(acc+bias); out=sigmoid(t+old)  -> bf16 in place (residual)
// blockIdx.z batches A/B/bias/C by the given element strides (grouped GEMM2).
// ---------------------------------------------------------------------------
template <int EPI>
__global__ __launch_bounds__(256) void gemm_bf16_kernel(
    const u16* __restrict__ A, int lda, long long sAz,
    const u16* __restrict__ B, int ldb, long long sBz,
    const float* __restrict__ bias, long long sBiasz,
    void* __restrict__ C, int ldc, long long sCz,
    int M, int N, int K)
{
    constexpr int BM = 128, BN = 128, BK = 64, LDT = 72; // LDS row stride (pad)
    constexpr int REG = BM * LDT;            // u16 elems per A (or B) region
    constexpr unsigned BUFB = 2u * REG * 2u; // bytes per double-buffer step
    __shared__ u16 smem[2 * 2 * REG];        // [buf][A|B][BM*LDT]

    const int tid  = threadIdx.x;
    const int wave = tid >> 5;
    const int lane = tid & 31;
    const int lm   = lane & 15;
    const int kh   = lane >> 4;      // lane half selects K sub-chunk
    const int wm   = wave & 3;       // 4 waves along M (32 rows each)
    const int wn   = wave >> 2;      // 2 waves along N (64 cols each)

    const int bm = blockIdx.y * BM;
    const int bn = blockIdx.x * BN;

    const u16*   Ab = A + (size_t)blockIdx.z * sAz;
    const u16*   Bb = B + (size_t)blockIdx.z * sBz;
    const float* bb = bias + (size_t)blockIdx.z * sBiasz;

    // ---- zero both LDS buffers once (covers rows masked off later) ----
    {
        uint4 z = make_uint4(0, 0, 0, 0);
        for (int j = tid * 8; j < 2 * 2 * REG; j += 256 * 8)
            *(uint4*)(&smem[j]) = z;
    }

    // ---- staging: 4 passes x (A row, B row), 16B per thread per pass ----
    const int ldr  = tid >> 3;           // 0..31
    const int ldc8 = (tid & 7) * 8;      // 0,8,...,56
    const u16* gA[4]; const u16* gB[4];
    unsigned   lA[4]; unsigned  lB[4];
    bool       vA[4]; bool      vB[4];
#pragma unroll
    for (int p = 0; p < 4; ++p) {
        int r = p * 32 + ldr;
        vA[p] = (bm + r) < M;
        vB[p] = (bn + r) < N;
        gA[p] = Ab + (size_t)(bm + r) * lda + ldc8;
        gB[p] = Bb + (size_t)(bn + r) * ldb + ldc8;
        lA[p] = (unsigned)(uintptr_t)&smem[r * LDT + ldc8];
        lB[p] = (unsigned)(uintptr_t)&smem[REG + r * LDT + ldc8];
    }
    const bool full = (bm + BM <= M) && (bn + BN <= N);  // block-uniform

    auto issue_full = [&](int buf, int kelem) {
        unsigned bo = (unsigned)buf * BUFB;
#pragma unroll
        for (int p = 0; p < 4; ++p) {
            async_load_b128(lA[p] + bo, gA[p] + kelem);
            async_load_b128(lB[p] + bo, gB[p] + kelem);
        }
    };
    auto issue_guard = [&](int buf, int kelem) {
        unsigned bo = (unsigned)buf * BUFB;
#pragma unroll
        for (int p = 0; p < 4; ++p) {
            if (vA[p]) async_load_b128(lA[p] + bo, gA[p] + kelem);
            if (vB[p]) async_load_b128(lB[p] + bo, gB[p] + kelem);
        }
    };

    v8f acc[2][4];
#pragma unroll
    for (int i = 0; i < 2; ++i)
#pragma unroll
        for (int j = 0; j < 4; ++j) acc[i][j] = (v8f)0.0f;

    __syncthreads();          // zeros visible before async writes can land
    if (full) issue_full(0, 0); else issue_guard(0, 0);   // tile 0 -> buf 0

    const int nt = K / BK;
    for (int i = 0; i < nt; ++i) {
        const int b = i & 1;
        wait_async0();        // this wave's portion of tile i has landed
        __syncthreads();      // all portions landed; buf 1-b reads all done
        if (i + 1 < nt) {
            if (full) issue_full(1 - b, (i + 1) * BK);
            else      issue_guard(1 - b, (i + 1) * BK);
        }

        const u16* As = &smem[b * 2 * REG];
        const u16* Bs = As + REG;

#pragma unroll
        for (int ks = 0; ks < BK; ks += 32) {
            v16bf af[2], bf[4];
#pragma unroll
            for (int im = 0; im < 2; ++im) {
                int ar = (wm * 32 + im * 16 + lm) * LDT + ks;
                // A 16x32 bf16: lane half kh -> K chunks [kh*8,+8) and [16+kh*8,+8)
                af[im] = make_frag(&As[ar + kh * 8], &As[ar + 16 + kh * 8]);
            }
#pragma unroll
            for (int in = 0; in < 4; ++in) {
                int br = (wn * 64 + in * 16 + lm) * LDT + ks;
                // B 32x16 bf16: lane half kh -> K chunks [kh*8,+8) and [16+kh*8,+8)
                bf[in] = make_frag(&Bs[br + kh * 8], &Bs[br + 16 + kh * 8]);
            }
#pragma unroll
            for (int im = 0; im < 2; ++im)
#pragma unroll
                for (int in = 0; in < 4; ++in)
                    acc[im][in] = __builtin_amdgcn_wmma_f32_16x16x32_bf16(
                        false, af[im], false, bf[in], (short)0, acc[im][in],
                        false, false);
        }
    }

    // ---------------- epilogue ----------------
#pragma unroll
    for (int im = 0; im < 2; ++im) {
#pragma unroll
        for (int in = 0; in < 4; ++in) {
            int col = bn + wn * 64 + in * 16 + lm;
            if (col >= N) continue;
            float bsv = bb[col];
#pragma unroll
            for (int r = 0; r < 8; ++r) {
                int row = bm + wm * 32 + im * 16 + r + kh * 8;
                if (row >= M) continue;
                float v = acc[im][in][r] + bsv;
                if constexpr (EPI == 0) {
                    float* Co = (float*)C + (size_t)blockIdx.z * sCz;
                    Co[(size_t)row * ldc + col] = v;
                } else if constexpr (EPI == 1) {
                    u16* Co = (u16*)C + (size_t)blockIdx.z * sCz;
                    Co[(size_t)row * ldc + col] = f2bf(sigmoidf_(v));
                } else if constexpr (EPI == 2) {
                    u16* Co = (u16*)C;
                    int g = col / GW, j = col - g * GW;
                    Co[(size_t)row * ldc + g * GW_PAD + j] = f2bf(sigmoidf_(v));
                } else { // EPI == 3: residual double-sigmoid, in place on h
                    u16* Co = (u16*)C + (size_t)blockIdx.z * sCz;
                    size_t o = (size_t)row * ldc + col;
                    float hold = bf2f(Co[o]);
                    float t = sigmoidf_(v);
                    Co[o] = f2bf(sigmoidf_(t + hold));
                }
            }
        }
    }
}

// ---------------------------------------------------------------------------
// Host launcher
// ---------------------------------------------------------------------------
static inline size_t align256(size_t x) { return (x + 255) & ~(size_t)255; }
static inline int cdiv(long long a, long long b) { return (int)((a + b - 1) / b); }

extern "C" void kernel_launch(void* const* d_in, const int* in_sizes, int n_in,
                              void* d_out, int out_size, void* d_ws, size_t ws_size,
                              hipStream_t stream) {
    const float* x      = (const float*)d_in[0];
    const float* mw     = (const float*)d_in[1];
    const float* mb     = (const float*)d_in[2];
    const float* h0_w   = (const float*)d_in[3];
    const float* h0_b   = (const float*)d_in[4];
    const float* bn_w1  = (const float*)d_in[5];
    const float* bn_b1  = (const float*)d_in[6];
    const float* bn_w2  = (const float*)d_in[7];
    const float* bn_b2  = (const float*)d_in[8];
    const float* out_w  = (const float*)d_in[9];
    const float* out_b  = (const float*)d_in[10];

    char* ws = (char*)d_ws;
    size_t off = 0;
    u16* m_bf   = (u16*)(ws + off); off += align256((size_t)BATCH * IN_PAD * 2);
    u16* h0w_bf = (u16*)(ws + off); off += align256((size_t)HIDDEN_W * IN_PAD * 2);
    u16* h_bf   = (u16*)(ws + off); off += align256((size_t)BATCH * KH_PAD * 2);
    u16* w1t_bf = (u16*)(ws + off); off += align256((size_t)HIDDEN_SIZE * HIDDEN_W * KH_PAD * 2);
    u16* w2t_bf = (u16*)(ws + off); off += align256((size_t)HIDDEN_SIZE * GROUP * GW * GW_PAD * 2);
    u16* g1_bf  = (u16*)(ws + off); off += align256((size_t)BATCH * GROUP * GW_PAD * 2);
    u16* owt_bf = (u16*)(ws + off); off += align256((size_t)OUT_DIM * KH_PAD * 2);
    (void)ws_size; (void)n_in; (void)in_sizes; (void)out_size;

    // 1. merge + bf16 conversions (+ zero pads that GEMMs read as K)
    merge_kernel<<<cdiv((long long)BATCH * IN_PAD, 256), 256, 0, stream>>>(x, mw, mb, m_bf);
    conv_h0w_kernel<<<cdiv((long long)HIDDEN_W * IN_PAD, 256), 256, 0, stream>>>(h0_w, h0w_bf);
    conv_w1_kernel<<<cdiv((long long)HIDDEN_SIZE * HIDDEN_W * KH_PAD, 256), 256, 0, stream>>>(bn_w1, w1t_bf);
    conv_w2_kernel<<<cdiv((long long)HIDDEN_SIZE * GROUP * GW * GW_PAD, 256), 256, 0, stream>>>(bn_w2, w2t_bf);
    conv_outw_kernel<<<cdiv((long long)OUT_DIM * KH_PAD, 256), 256, 0, stream>>>(out_w, owt_bf);
    zero_u16_kernel<<<cdiv((long long)BATCH * GROUP * GW_PAD, 256), 256, 0, stream>>>(
        g1_bf, BATCH * GROUP * GW_PAD);
    zero_u16_kernel<<<cdiv((long long)BATCH * KH_PAD, 256), 256, 0, stream>>>(
        h_bf, BATCH * KH_PAD);

    // 2. h0:  h = sigmoid(m @ h0_w^T + b)      [1024,2000], K=5056
    gemm_bf16_kernel<1><<<dim3(cdiv(HIDDEN_W, 128), cdiv(BATCH, 128), 1), 256, 0, stream>>>(
        m_bf, IN_PAD, 0, h0w_bf, IN_PAD, 0, h0_b, 0,
        h_bf, KH_PAD, 0, BATCH, HIDDEN_W, IN_PAD);

    // 3. bottleneck layers
    for (int l = 0; l < HIDDEN_SIZE; ++l) {
        // g1 = sigmoid(h @ w1t^T + b1) -> group-padded [B, G, 128], K=2048
        gemm_bf16_kernel<2><<<dim3(cdiv(HIDDEN_W, 128), cdiv(BATCH, 128), 1), 256, 0, stream>>>(
            h_bf, KH_PAD, 0,
            w1t_bf + (size_t)l * HIDDEN_W * KH_PAD, KH_PAD, 0,
            bn_b1 + (size_t)l * HIDDEN_W, 0,
            g1_bf, GROUP * GW_PAD, 0, BATCH, HIDDEN_W, KH_PAD);
        // h = sigmoid(sigmoid(g1 @ w2t^T + b2) + h)   grouped over z=20, K=128
        gemm_bf16_kernel<3><<<dim3(1, cdiv(BATCH, 128), GROUP), 256, 0, stream>>>(
            g1_bf, GROUP * GW_PAD, GW_PAD,
            w2t_bf + (size_t)l * GROUP * GW * GW_PAD, GW_PAD, (long long)GW * GW_PAD,
            bn_b2 + (size_t)l * GROUP * GW, GW,
            h_bf, KH_PAD, GW, BATCH, GW, GW_PAD);
    }

    // 4. out = h @ out_w^T + b   -> f32 [1024, 200], K=2048
    gemm_bf16_kernel<0><<<dim3(cdiv(OUT_DIM, 128), cdiv(BATCH, 128), 1), 256, 0, stream>>>(
        h_bf, KH_PAD, 0, owt_bf, KH_PAD, 0, out_b, 0,
        d_out, OUT_DIM, 0, BATCH, OUT_DIM, KH_PAD);
}
